// Attention_org_45767171506399
// MI455X (gfx1250) — matrix operand order repaired and checked by reading, not compiled, not run on hardware
//
#include <hip/hip_runtime.h>
#include <hip/hip_bf16.h>
#include <math.h>

typedef __attribute__((ext_vector_type(16))) __bf16 bf16x16;
typedef __attribute__((ext_vector_type(8)))  float  f32x8;

#define TILE_M 128
#define TILE_N 128
#define TILE_K 32
#define LDS_PITCH 40   // bf16 elems per LDS row: 32 data + 8 pad (80B, conflict-free)
#define BUF_BYTES (TILE_M * LDS_PITCH * 2)   // 10240 bytes per tile buffer

__device__ __forceinline__ unsigned short f32_to_bf16_rn(float f) {
    unsigned int u = __float_as_uint(f);
    unsigned int r = (u + 0x7FFFu + ((u >> 16) & 1u)) >> 16;
    return (unsigned short)r;
}

__device__ __forceinline__ void store_out(float* p, float v) { *p = v; }
__device__ __forceinline__ void store_out(unsigned short* p, float v) { *p = f32_to_bf16_rn(v); }

// gfx1250 async global->LDS copy (ASYNCcnt-tracked, no VGPR staging)
__device__ __forceinline__ void async_copy_b128(unsigned lds_byte_off, const void* gptr) {
    asm volatile("global_load_async_to_lds_b128 %0, %1, off"
                 :: "v"(lds_byte_off), "v"(gptr) : "memory");
}
__device__ __forceinline__ void wait_async0() {
    asm volatile("s_wait_asynccnt 0x0" ::: "memory");
}

union AFrag { bf16x16 v; uint4 q[2]; };

// C[m,n] = scale * sum_inner sum_k A[m,k] * B[n,k]
// A: [batch][M][K] bf16 (ld=K), B: [batch][N][K] bf16 (ld=K), C: [batch][M][N]
// 128 threads = 4 waves; block tile 128x128; wave tile 64x64 (16 WMMA accs).
template <typename OutT>
__global__ __launch_bounds__(128) void gemm_bf16_nt(
    const unsigned short* __restrict__ A,
    const unsigned short* __restrict__ B,
    OutT* __restrict__ C,
    int M, int N, int K,
    int batchH,
    long aStrideB, long aStrideH,
    long bStrideB, long bStrideH,
    long cStrideB, long cStrideH,
    int innerCount, long aStrideInner, long bStrideInner,
    float scale)
{
    __shared__ __align__(16) unsigned short lA[2][TILE_M * LDS_PITCH];
    __shared__ __align__(16) unsigned short lB[2][TILE_N * LDS_PITCH];

    const int tid  = threadIdx.x;
    const int lane = tid & 31;
    const int wid  = tid >> 5;     // 0..3 waves
    const int wm   = wid >> 1;     // 0..1 : 64-row group
    const int wn   = wid & 1;      // 0..1 : 64-col group
    const int hs   = lane >> 4;    // half-wave select
    const int lm   = lane & 15;

    const int zb = (int)blockIdx.z;
    const int b  = zb / batchH;
    const int h  = zb - b * batchH;

    const unsigned short* Ab = A + (long)b * aStrideB + (long)h * aStrideH;
    const unsigned short* Bb = B + (long)b * bStrideB + (long)h * bStrideH;
    OutT* Cb = C + (long)b * cStrideB + (long)h * cStrideH;

    const int m0 = (int)blockIdx.y * TILE_M;
    const int n0 = (int)blockIdx.x * TILE_N;

    // Staging geometry: 512 16B chunks per matrix tile, 4 per thread per matrix.
    // Keep live 64-bit global pointers; advance incrementally (no per-step muls).
    const unsigned short* gA[4];
    const unsigned short* gB[4];
    unsigned ldsAoff[4], ldsBoff[4];
    #pragma unroll
    for (int i = 0; i < 4; ++i) {
        int c   = tid + i * 128;       // 0..511
        int row = c >> 2;              // 0..127
        int ko  = (c & 3) * 8;         // 0,8,16,24 (bf16 elems)
        int gm = m0 + row; gm = (gm < M) ? gm : (M - 1);
        int gn = n0 + row; gn = (gn < N) ? gn : (N - 1);
        gA[i] = Ab + (long)gm * K + ko;
        gB[i] = Bb + (long)gn * K + ko;
        ldsAoff[i] = (unsigned)(unsigned long long)&lA[0][row * LDS_PITCH + ko];
        ldsBoff[i] = (unsigned)(unsigned long long)&lB[0][row * LDS_PITCH + ko];
    }

    auto stage = [&](int buf) {
        const unsigned addb = buf ? (unsigned)BUF_BYTES : 0u;
        #pragma unroll
        for (int i = 0; i < 4; ++i) {
            async_copy_b128(ldsAoff[i] + addb, gA[i]);
            async_copy_b128(ldsBoff[i] + addb, gB[i]);
        }
    };
    auto advance = [&](long dA, long dB) {
        #pragma unroll
        for (int i = 0; i < 4; ++i) { gA[i] += dA; gB[i] += dB; }
    };

    f32x8 zero = {0.f,0.f,0.f,0.f,0.f,0.f,0.f,0.f};
    f32x8 acc[4][4];
    #pragma unroll
    for (int i = 0; i < 4; ++i)
        #pragma unroll
        for (int j = 0; j < 4; ++j)
            acc[i][j] = zero;

    const int stepsPer = K >> 5;            // K is a multiple of 32
    const int total    = innerCount * stepsPer;

    // Prologue: stage step 0, then advance pointers to step 1.
    stage(0);
    {
        int kpos = TILE_K;
        long dA = TILE_K, dB = TILE_K;
        if (kpos >= K) { dA += aStrideInner - K; dB += bStrideInner - K; }
        advance(dA, dB);
    }
    int kpos = (TILE_K >= K) ? 0 : TILE_K;  // k-position of the step just pointed to

    for (int t = 0; t < total; ++t) {
        const int cur = t & 1;
        wait_async0();        // our async copies for buffer `cur` have landed in LDS
        __syncthreads();      // everyone's have; prior readers of buffer cur^1 are done
        if (t + 1 < total) {
            stage(cur ^ 1);   // overlap next tile copy with compute
            long dA = TILE_K, dB = TILE_K;
            kpos += TILE_K;
            if (kpos >= K) { kpos = 0; dA += aStrideInner - K; dB += bStrideInner - K; }
            advance(dA, dB);
        }

        // A fragment: lane holds row m=lm, K = hs*16 + {0..15} contiguous
        AFrag af[4];
        #pragma unroll
        for (int tm = 0; tm < 4; ++tm) {
            const unsigned short* src = &lA[cur][(wm * 64 + tm * 16 + lm) * LDS_PITCH + hs * 16];
            af[tm].q[0] = *(const uint4*)(src);
            af[tm].q[1] = *(const uint4*)(src + 8);
        }
        // B fragment: lane holds col n=lm, K = hs*16 + {0..15} contiguous
        AFrag bfr[4];
        #pragma unroll
        for (int tn = 0; tn < 4; ++tn) {
            const unsigned short* src = &lB[cur][(wn * 64 + tn * 16 + lm) * LDS_PITCH + hs * 16];
            bfr[tn].q[0] = *(const uint4*)(src);
            bfr[tn].q[1] = *(const uint4*)(src + 8);
        }
        #pragma unroll
        for (int tm = 0; tm < 4; ++tm)
            #pragma unroll
            for (int tn = 0; tn < 4; ++tn)
                acc[tm][tn] = __builtin_amdgcn_wmma_f32_16x16x32_bf16(
                    false, af[tm].v, false, bfr[tn].v, (short)0, acc[tm][tn], false, false);
    }

    // Epilogue: C/D layout -> VGPR r: row = r + hs*8, col = lm
    #pragma unroll
    for (int tm = 0; tm < 4; ++tm)
        #pragma unroll
        for (int tn = 0; tn < 4; ++tn)
            #pragma unroll
            for (int r = 0; r < 8; ++r) {
                int gm = m0 + wm * 64 + tm * 16 + r + hs * 8;
                int gn = n0 + wn * 64 + tn * 16 + lm;
                if (gm < M && gn < N)
                    store_out(&Cb[(long)gm * N + gn], acc[tm][tn][r] * scale);
            }
}

__global__ void cvt_f32_bf16(const float* __restrict__ in, unsigned short* __restrict__ out, long n) {
    long i = (long)blockIdx.x * blockDim.x + threadIdx.x;
    long stride = (long)gridDim.x * blockDim.x;
    for (; i < n; i += stride) out[i] = f32_to_bf16_rn(in[i]);
}

// per (b,h): mean / rsqrt(var+eps) over C*KV elements
__global__ __launch_bounds__(256) void inst_stats(const float* __restrict__ s,
                                                  float* __restrict__ stats, long elems) {
    long base = (long)blockIdx.x * elems;
    float sum = 0.f, ss = 0.f;
    for (long i = threadIdx.x; i < elems; i += 256) {
        float v = s[base + i]; sum += v; ss += v * v;
    }
    __shared__ float rs[256], rq[256];
    rs[threadIdx.x] = sum; rq[threadIdx.x] = ss;
    __syncthreads();
    for (int st = 128; st > 0; st >>= 1) {
        if (threadIdx.x < st) { rs[threadIdx.x] += rs[threadIdx.x + st];
                                rq[threadIdx.x] += rq[threadIdx.x + st]; }
        __syncthreads();
    }
    if (threadIdx.x == 0) {
        float inv = 1.f / (float)elems;
        float mu  = rs[0] * inv;
        float var = rq[0] * inv - mu * mu;
        stats[blockIdx.x * 2 + 0] = mu;
        stats[blockIdx.x * 2 + 1] = rsqrtf(var + 1e-5f);
    }
}

// row-wise: normalize then softmax, write bf16 p
__global__ __launch_bounds__(256) void norm_softmax(
    const float* __restrict__ s, unsigned short* __restrict__ p,
    const float* __restrict__ stats, int C, int KV)
{
    int d = blockIdx.x, bh = blockIdx.y;
    long rowoff = ((long)bh * C + d) * (long)KV;
    float mu = stats[bh * 2], rsig = stats[bh * 2 + 1];
    float v[4]; int cnt = 0;
    float mx = -3.402823e38f;
    for (int i = threadIdx.x, j = 0; i < KV; i += 256, ++j) {
        float x = (s[rowoff + i] - mu) * rsig;
        v[j] = x; mx = fmaxf(mx, x); cnt = j + 1;
    }
    __shared__ float red[256];
    red[threadIdx.x] = mx; __syncthreads();
    for (int st = 128; st > 0; st >>= 1) {
        if (threadIdx.x < st) red[threadIdx.x] = fmaxf(red[threadIdx.x], red[threadIdx.x + st]);
        __syncthreads();
    }
    mx = red[0]; __syncthreads();
    float sum = 0.f;
    for (int j = 0; j < cnt; ++j) { v[j] = __expf(v[j] - mx); sum += v[j]; }
    red[threadIdx.x] = sum; __syncthreads();
    for (int st = 128; st > 0; st >>= 1) {
        if (threadIdx.x < st) red[threadIdx.x] += red[threadIdx.x + st];
        __syncthreads();
    }
    float inv = 1.f / red[0];
    for (int i = threadIdx.x, j = 0; i < KV; i += 256, ++j)
        p[rowoff + i] = f32_to_bf16_rn(v[j] * inv);
}

extern "C" void kernel_launch(void* const* d_in, const int* in_sizes, int n_in,
                              void* d_out, int out_size, void* d_ws, size_t ws_size,
                              hipStream_t stream) {
    (void)in_sizes; (void)n_in; (void)out_size; (void)ws_size;
    const int Bn = 8, Nn = 1024, Hn = 4, KVn = 960;
    const int CH[3] = {160, 320, 480};

    const float* emb[3] = {(const float*)d_in[0], (const float*)d_in[1], (const float*)d_in[2]};
    const float* embAll = (const float*)d_in[3];
    const float* Wq[3]  = {(const float*)d_in[4], (const float*)d_in[5], (const float*)d_in[6]};
    const float* Wk     = (const float*)d_in[7];
    const float* Wv     = (const float*)d_in[8];
    const float* Wo[3]  = {(const float*)d_in[9], (const float*)d_in[10], (const float*)d_in[11]};

    // ---- workspace layout ----
    char* base = (char*)d_ws;
    size_t off = 0;
    auto alloc = [&](size_t bytes) -> char* {
        off = (off + 255) & ~(size_t)255;
        char* r = base + off; off += bytes; return r;
    };
    unsigned short* embAll_bf = (unsigned short*)alloc((size_t)Bn * Nn * KVn * 2);
    unsigned short* emb_bf[3], *Wq_bf[3], *Wo_bf[3];
    for (int i = 0; i < 3; ++i) emb_bf[i] = (unsigned short*)alloc((size_t)Bn * Nn * CH[i] * 2);
    for (int i = 0; i < 3; ++i) Wq_bf[i]  = (unsigned short*)alloc((size_t)Hn * CH[i] * CH[i] * 2);
    unsigned short* Wk_bf = (unsigned short*)alloc((size_t)Hn * KVn * KVn * 2);
    unsigned short* Wv_bf = (unsigned short*)alloc((size_t)Hn * KVn * KVn * 2);
    for (int i = 0; i < 3; ++i) Wo_bf[i]  = (unsigned short*)alloc((size_t)CH[i] * CH[i] * 2);
    unsigned short* Kt  = (unsigned short*)alloc((size_t)Bn * Hn * KVn * Nn * 2); // [b][h][e][n]
    unsigned short* Vb  = (unsigned short*)alloc((size_t)Bn * Hn * Nn * KVn * 2); // [b][h][n][e]
    unsigned short* Qt  = (unsigned short*)alloc((size_t)Bn * Hn * 480 * Nn * 2); // [b][h][d][n]
    float*          sb  = (float*)alloc((size_t)Bn * Hn * 480 * KVn * 4);         // [b][h][d][e]
    unsigned short* ctx = (unsigned short*)alloc((size_t)Bn * Nn * 480 * 2);      // [b][n][d]
    float*        stats = (float*)alloc(32 * 2 * 4);
    unsigned short* pbf = Qt;  // p reuses Qt storage (Qt dead once s is computed)

    auto cvt = [&](const float* in, unsigned short* out, long n) {
        int blocks = (int)((n + 1023) / 1024); if (blocks > 4096) blocks = 4096;
        cvt_f32_bf16<<<blocks, 256, 0, stream>>>(in, out, n);
    };
    cvt(embAll, embAll_bf, (long)Bn * Nn * KVn);
    cvt(Wk, Wk_bf, (long)Hn * KVn * KVn);
    cvt(Wv, Wv_bf, (long)Hn * KVn * KVn);
    for (int i = 0; i < 3; ++i) {
        cvt(emb[i], emb_bf[i], (long)Bn * Nn * CH[i]);
        cvt(Wq[i],  Wq_bf[i],  (long)Hn * CH[i] * CH[i]);
        cvt(Wo[i],  Wo_bf[i],  (long)CH[i] * CH[i]);
    }

    auto grid_of = [](int M, int N, int batch) { return dim3((N + 127) / 128, (M + 127) / 128, batch); };

    // 1) Kt[b,h][e][n] = sum_k Wk[h][e,k] * embAll[b][n,k]
    gemm_bf16_nt<unsigned short><<<grid_of(KVn, Nn, 32), 128, 0, stream>>>(
        Wk_bf, embAll_bf, Kt, KVn, Nn, KVn, Hn,
        0L, (long)KVn * KVn, (long)Nn * KVn, 0L,
        (long)Hn * KVn * Nn, (long)KVn * Nn, 1, 0L, 0L, 1.0f);
    // 2) V[b,h][n][e] = sum_k embAll[b][n,k] * Wv[h][e,k]
    gemm_bf16_nt<unsigned short><<<grid_of(Nn, KVn, 32), 128, 0, stream>>>(
        embAll_bf, Wv_bf, Vb, Nn, KVn, KVn, Hn,
        (long)Nn * KVn, 0L, 0L, (long)KVn * KVn,
        (long)Hn * Nn * KVn, (long)Nn * KVn, 1, 0L, 0L, 1.0f);

    long outOff = 0;
    for (int br = 0; br < 3; ++br) {
        const int Cc = CH[br];
        // 3) Qt[b,h][d][n] = sum_c Wq[h][d,c] * emb[b][n,c]
        gemm_bf16_nt<unsigned short><<<grid_of(Cc, Nn, 32), 128, 0, stream>>>(
            Wq_bf[br], emb_bf[br], Qt, Cc, Nn, Cc, Hn,
            0L, (long)Cc * Cc, (long)Nn * Cc, 0L,
            (long)Hn * Cc * Nn, (long)Cc * Nn, 1, 0L, 0L, 1.0f);
        // 4) s[b,h][d][e] = (1/sqrt(KV)) sum_n Qt[d,n] * Kt[e,n]
        gemm_bf16_nt<float><<<grid_of(Cc, KVn, 32), 128, 0, stream>>>(
            Qt, Kt, sb, Cc, KVn, Nn, Hn,
            (long)Hn * Cc * Nn, (long)Cc * Nn, (long)Hn * KVn * Nn, (long)KVn * Nn,
            (long)Hn * Cc * KVn, (long)Cc * KVn, 1, 0L, 0L, 1.0f / sqrtf((float)KVn));
        // 5) instance-norm stats + fused normalize/softmax -> p (bf16, reuses Qt)
        inst_stats<<<dim3(32), 256, 0, stream>>>(sb, stats, (long)Cc * KVn);
        norm_softmax<<<dim3(Cc, 32), 256, 0, stream>>>(sb, pbf, stats, Cc, KVn);
        // 6) ctx[b][n][d] = (1/H) sum_h sum_e V[b,h][n,e] * p[b,h][d,e]
        gemm_bf16_nt<unsigned short><<<grid_of(Nn, Cc, 8), 128, 0, stream>>>(
            Vb, pbf, ctx, Nn, Cc, KVn, 1,
            (long)Hn * Nn * KVn, 0L, (long)Hn * Cc * KVn, 0L,
            (long)Nn * Cc, 0L, Hn, (long)Nn * KVn, (long)Cc * KVn, 1.0f / Hn);
        // 7) out[b][n][c] = sum_d ctx[b][n,d] * Wo[c,d]
        gemm_bf16_nt<float><<<grid_of(Nn, Cc, 8), 128, 0, stream>>>(
            ctx, Wo_bf[br], (float*)d_out + outOff, Nn, Cc, Cc, 1,
            (long)Nn * Cc, 0L, 0L, 0L,
            (long)Nn * Cc, 0L, 1, 0L, 0L, 1.0f);
        outOff += (long)Bn * Nn * Cc;
    }
}
